// RUMGraphRegressionModel_65824668778862
// MI455X (gfx1250) — matrix operand order repaired and checked by reading, not compiled, not run on hardware
//
#include <hip/hip_runtime.h>
#include <hip/hip_bf16.h>
#include <math.h>

// ---------------- problem constants (from reference) ----------------
#define D_HID   128
#define F_IN    64
#define S_WALK  4
#define L_WALK  8
#define N_NODES 100000
#define B_GRAPH 256
#define HEADS   8
#define SN      (S_WALK * N_NODES)          // 400000, divisible by 16
#define SS_W    0.05f

typedef __attribute__((ext_vector_type(16))) _Float16 v16h;
typedef __attribute__((ext_vector_type(8)))  float    v8f;

// ---------------- WMMA helpers ----------------
__device__ __forceinline__ v8f wmma16(v16h a, v16h b, v8f c) {
  // D = A(16x32 f16) * B(32x16 f16) + C(16x16 f32)
  return __builtin_amdgcn_wmma_f32_16x16x32_f16(false, a, false, b, (short)0, c,
                                                false, false);
}

__device__ __forceinline__ v8f splat8(float x) {
  v8f r;
#pragma unroll
  for (int i = 0; i < 8; ++i) r[i] = x;
  return r;
}

// Load one 16x32 f16 A tile from an f32 row-major source.
// ISA layout (16-bit A 16x32): lane M = lane&15, hi = lane>>4;
// VGPR v: K = (v>>2)*16 + hi*8 + 2*(v&3) (+1 for odd half).
__device__ __forceinline__ v16h load_a_f32(const float* __restrict__ row,
                                           int kbase, int hi) {
  v16h a;
#pragma unroll
  for (int v = 0; v < 8; ++v) {
    int k = kbase + ((v >> 2) << 4) + (hi << 3) + ((v & 3) << 1);
    float2 f = *(const float2*)(row + k);
    a[2 * v]     = (_Float16)f.x;
    a[2 * v + 1] = (_Float16)f.y;
  }
  return a;
}

// Pre-packed B tile: 512 halfs per 32x16 tile, lane-major (lane*16 + e),
// lane col = lane&15, K = (e>>3)*16 + (lane>>4)*8 + (e&7).  Two b128 loads.
__device__ __forceinline__ v16h ld_btile(const _Float16* base, int tileIdx,
                                         int lane) {
  return *(const v16h*)(base + (size_t)tileIdx * 512 + lane * 16);
}

__device__ __forceinline__ float sigf(float x) { return 1.f / (1.f + __expf(-x)); }

// sortable-uint mapping for float atomicMax
__device__ __forceinline__ unsigned f2s(float f) {
  unsigned u = __float_as_uint(f);
  return (u & 0x80000000u) ? ~u : (u | 0x80000000u);
}
__device__ __forceinline__ float s2f(unsigned s) {
  unsigned u = (s & 0x80000000u) ? (s & 0x7fffffffu) : ~s;
  return __uint_as_float(u);
}

// ---------------- CDNA5 async global->LDS copy (ASYNCcnt path) ---------------
// Recover the 32-bit LDS byte offset of a generic pointer into shared memory.
__device__ __forceinline__ unsigned lds_off_u32(const void* p) {
  return (unsigned)(unsigned long long)
      (__attribute__((address_space(3))) const char*)p;
}
// GLOBAL_LOAD_ASYNC_TO_LDS_B128: per-lane 16B copy, tracked with ASYNCcnt.
__device__ __forceinline__ void async_g2l_b128(unsigned lds_byte_off,
                                               const void* gaddr) {
  asm volatile("global_load_async_to_lds_b128 %0, %1, off"
               :: "v"(lds_byte_off), "v"(gaddr)
               : "memory");
}
__device__ __forceinline__ void wait_asynccnt0() {
  asm volatile("s_wait_asynccnt 0x0" ::: "memory");
}

// ---------------- weight packing: f32 [K,N] -> WMMA B tiles f16 ----------------
__global__ void rum_pack_b(const float* __restrict__ W, _Float16* __restrict__ out,
                           int K, int Ncols) {
  int numKt = K >> 5;
  int numNt = (Ncols + 15) >> 4;
  int total = numKt * numNt * 512;
  for (int id = blockIdx.x * blockDim.x + threadIdx.x; id < total;
       id += gridDim.x * blockDim.x) {
    int tile = id >> 9;
    int e    = id & 511;
    int lane = e >> 4;
    int el   = e & 15;
    int nt = tile / numKt;
    int kt = tile % numKt;
    int k   = kt * 32 + (el >> 3) * 16 + (lane >> 4) * 8 + (el & 7);
    int col = nt * 16 + (lane & 15);
    float v = (col < Ncols) ? W[(size_t)k * Ncols + col] : 0.f;
    out[id] = (_Float16)v;
  }
}

// ---------------- accumulator init ----------------
__global__ void rum_init(float* loss, unsigned* m, float* z, float* outp) {
  int i = blockIdx.x * blockDim.x + threadIdx.x;
  if (i == 0) *loss = 0.f;
  if (i < B_GRAPH) { m[i] = 0u; z[i] = 0.f; }
  if (i < B_GRAPH * D_HID) outp[i] = 0.f;
}

// ------- generic WMMA GEMM: C = act(A[M, NKT*32] @ B + bias), compile-time ----
// Template params keep every register-array index constant (no v_movrels).
template <int NKT, int NCOLS, int ACT>
__global__ __launch_bounds__(128)
void rum_gemm(const float* __restrict__ A, const _Float16* __restrict__ Bp,
              const float* __restrict__ bias, float* __restrict__ C, int Mtot) {
  constexpr int K   = NKT * 32;
  constexpr int NNT = (NCOLS + 15) / 16;
  constexpr int TILES = NKT * NNT;
  extern __shared__ char dynsmem[];
  _Float16* bs = (_Float16*)dynsmem;
  {
    int4* dst = (int4*)dynsmem;
    const int4* src = (const int4*)Bp;
    for (int i = threadIdx.x; i < TILES * 64; i += blockDim.x) dst[i] = src[i];
  }
  __syncthreads();
  const int wid  = threadIdx.x >> 5;
  const int lane = threadIdx.x & 31;
  const int li   = lane & 15;
  const int hi   = lane >> 4;
  const int mtiles = Mtot >> 4;
  for (int mt = blockIdx.x * 4 + wid; mt < mtiles; mt += gridDim.x * 4) {
    const float* arow = A + (size_t)(mt * 16 + li) * K;
    v16h a[NKT];
#pragma unroll
    for (int kt = 0; kt < NKT; ++kt) a[kt] = load_a_f32(arow, kt * 32, hi);
#pragma unroll
    for (int nt = 0; nt < NNT; ++nt) {
      int col = nt * 16 + li;
      v8f acc = splat8(0.f);              // zero init; bias folded in epilogue
#pragma unroll
      for (int kt = 0; kt < NKT; ++kt)
        acc = wmma16(a[kt], ld_btile(bs, nt * NKT + kt, lane), acc);
      if (col < NCOLS) {
        float bv = bias[col];
#pragma unroll
        for (int v = 0; v < 8; ++v) {
          float x = acc[v] + bv;
          if (ACT == 1) x = x * sigf(x);       // SiLU
          C[(size_t)(mt * 16 + v + hi * 8) * NCOLS + col] = x;
        }
      }
    }
  }
}

// ---------------- fused RUM GRU layer (L steps, state in registers) -----------
// 256 threads = 8 waves/block; one block per WGP (256 KB LDS) -> 2 waves/SIMD
// so LDS latency of B-tile loads is hidden by the co-resident wave.
// All WMMA accumulators start at zero (rematerializable) and the GRU biases are
// folded into the scalar gating tail -- keeps register pressure low enough to
// avoid scratch spills at 2 waves/SIMD.
__global__ __launch_bounds__(256, 1)
void rum_gru_layer(const float* __restrict__ hin, const int* __restrict__ walk,
                   const _Float16* __restrict__ WihP,
                   const _Float16* __restrict__ WhhP,
                   const float* __restrict__ bih, const float* __restrict__ bhh,
                   const _Float16* __restrict__ DecP, const float* __restrict__ db,
                   const float* __restrict__ h0, float* __restrict__ state,
                   float* __restrict__ loss_accum, float loss_scale) {
  extern __shared__ char dynsmem[];
  _Float16* wih_s = (_Float16*)dynsmem;              // 24*4*512 = 49152 halfs
  _Float16* whh_s = wih_s + 24 * 4 * 512;            // 49152 halfs
  float*    st_lds = (float*)(whh_s + 24 * 4 * 512); // 8 waves * 16*128 f32

  const int tid  = threadIdx.x;
  const int wid  = tid >> 5;
  const int lane = tid & 31;
  const int li   = lane & 15;
  const int hi   = lane >> 4;

  // Stage both weight matrices into LDS (192 KB) with async global->LDS DMA:
  // no VGPR round-trip, tracked on ASYNCcnt.
  {
    unsigned l0 = lds_off_u32(wih_s);
    unsigned l1 = lds_off_u32(whh_s);
    const char* g0 = (const char*)WihP;
    const char* g1 = (const char*)WhhP;
    for (int i = tid; i < 6144; i += 256) {          // 6144 x 16B per matrix
      unsigned b = (unsigned)i * 16u;
      async_g2l_b128(l0 + b, g0 + b);
      async_g2l_b128(l1 + b, g1 + b);
    }
    wait_asynccnt0();
  }
  __syncthreads();

  const int tile   = blockIdx.x * 8 + wid;      // 16-row tile of flattened [S*N]
  const int r_lane = tile * 16 + li;            // this lane's M-row
  float* myst = st_lds + wid * 16 * D_HID;

  v8f st[8];
#pragma unroll
  for (int j = 0; j < 8; ++j) st[j] = splat8(0.f);

  for (int l = 0; l < L_WALK; ++l) {
    // gather x = h[walk[.,l]] ; prefetch next step's row
    int idx = walk[(size_t)r_lane * L_WALK + l];
    if (l + 1 < L_WALK) {
      int nidx = walk[(size_t)r_lane * L_WALK + l + 1];
      __builtin_prefetch(hin + (size_t)nidx * D_HID, 0, 1);
    }
    const float* xrow = hin + (size_t)idx * D_HID;
    v16h ax[4], as[4];
#pragma unroll
    for (int kt = 0; kt < 4; ++kt) ax[kt] = load_a_f32(xrow, kt * 32, hi);

    // transpose register state (C layout) -> LDS -> A layout (f16)
    __syncthreads();
#pragma unroll
    for (int j = 0; j < 8; ++j)
#pragma unroll
      for (int v = 0; v < 8; ++v)
        myst[(v + hi * 8) * D_HID + j * 16 + li] = st[j][v];
    __syncthreads();
#pragma unroll
    for (int kt = 0; kt < 4; ++kt) as[kt] = load_a_f32(myst + li * D_HID, kt * 32, hi);

#pragma unroll
    for (int j = 0; j < 8; ++j) {
      int cr = j * 16 + li, cz = 128 + cr, cn = 256 + cr;
      v8f ar  = splat8(0.f);
      v8f az  = splat8(0.f);
      v8f ain = splat8(0.f);
      v8f ahn = splat8(0.f);
      int ntr = j, ntz = j + 8, ntn = j + 16;
#pragma unroll
      for (int kt = 0; kt < 4; ++kt) {
        // Batch the six independent B-tile loads so the scheduler can overlap
        // ds_load latency with the WMMA stream instead of draining DScnt.
        v16h b_ri = ld_btile(wih_s, ntr * 4 + kt, lane);
        v16h b_rh = ld_btile(whh_s, ntr * 4 + kt, lane);
        v16h b_zi = ld_btile(wih_s, ntz * 4 + kt, lane);
        v16h b_zh = ld_btile(whh_s, ntz * 4 + kt, lane);
        v16h b_ni = ld_btile(wih_s, ntn * 4 + kt, lane);
        v16h b_nh = ld_btile(whh_s, ntn * 4 + kt, lane);
        ar  = wmma16(ax[kt], b_ri, ar);
        ar  = wmma16(as[kt], b_rh, ar);
        az  = wmma16(ax[kt], b_zi, az);
        az  = wmma16(as[kt], b_zh, az);
        ain = wmma16(ax[kt], b_ni, ain);
        ahn = wmma16(as[kt], b_nh, ahn);
      }
      // biases folded here (scalar, cheap) instead of fat splat accumulators
      float br  = bih[cr] + bhh[cr];
      float bz  = bih[cz] + bhh[cz];
      float bni = bih[cn];
      float bnh = bhh[cn];
#pragma unroll
      for (int v = 0; v < 8; ++v) {
        float r = sigf(ar[v] + br);
        float z = sigf(az[v] + bz);
        float n = tanhf(ain[v] + bni + r * (ahn[v] + bnh));
        st[j][v] = (1.f - z) * n + z * st[j][v];
      }
    }
  }

  // final transpose for the decode matmul
  __syncthreads();
#pragma unroll
  for (int j = 0; j < 8; ++j)
#pragma unroll
    for (int v = 0; v < 8; ++v)
      myst[(v + hi * 8) * D_HID + j * 16 + li] = st[j][v];
  __syncthreads();
  v16h ah[4];
#pragma unroll
  for (int kt = 0; kt < 4; ++kt) ah[kt] = load_a_f32(myst + li * D_HID, kt * 32, hi);

  // store hT to global state [S*N, D]
#pragma unroll
  for (int j = 0; j < 8; ++j)
#pragma unroll
    for (int v = 0; v < 8; ++v)
      state[(size_t)(tile * 16 + v + hi * 8) * D_HID + j * 16 + li] = st[j][v];

  // rec = hT @ dec_w + db ; loss += scale * sum((rec - h0)^2)
  float sq = 0.f;
#pragma unroll
  for (int nt = 0; nt < 4; ++nt) {
    int col = nt * 16 + li;                       // col < 64 always
    v8f acc = splat8(0.f);
#pragma unroll
    for (int kt = 0; kt < 4; ++kt)
      acc = wmma16(ah[kt], ld_btile(DecP, nt * 4 + kt, lane), acc);
    float bd = db[col];
#pragma unroll
    for (int v = 0; v < 8; ++v) {
      int r = tile * 16 + v + hi * 8;
      int n = r % N_NODES;
      float e = acc[v] + bd - h0[(size_t)n * F_IN + col];
      sq += e * e;
    }
  }
  for (int off = 16; off; off >>= 1) sq += __shfl_down(sq, off);
  if (lane == 0) atomicAdd(loss_accum, sq * loss_scale);
}

// ---------------- elementwise tails ----------------
__global__ void rum_silu_mean(const float* __restrict__ st, float* __restrict__ out,
                              int total) {
  int i = blockIdx.x * blockDim.x + threadIdx.x;
  if (i >= total) return;
  float acc = 0.f;
#pragma unroll
  for (int s = 0; s < S_WALK; ++s) {
    float x = st[(size_t)s * total + i];
    acc += x * sigf(x);
  }
  out[i] = acc * (1.f / S_WALK);
}

__global__ void rum_layernorm(float* __restrict__ h, const float* __restrict__ g,
                              const float* __restrict__ b, int rows) {
  int row = (blockIdx.x * blockDim.x + threadIdx.x) >> 5;
  int lane = threadIdx.x & 31;
  if (row >= rows) return;
  float* rp = h + (size_t)row * D_HID;
  float x[4], s = 0.f, s2 = 0.f;
#pragma unroll
  for (int k = 0; k < 4; ++k) {
    x[k] = rp[lane + 32 * k];
    s += x[k];
    s2 += x[k] * x[k];
  }
  for (int off = 16; off; off >>= 1) {
    s  += __shfl_xor(s, off);
    s2 += __shfl_xor(s2, off);
  }
  float mu  = s * (1.f / D_HID);
  float var = s2 * (1.f / D_HID) - mu * mu;
  float inv = rsqrtf(var + 1e-5f);
#pragma unroll
  for (int k = 0; k < 4; ++k) {
    int d = lane + 32 * k;
    rp[d] = (x[k] - mu) * inv * g[d] + b[d];
  }
}

__global__ void rum_mha_combine(const float* __restrict__ f,
                                const float* __restrict__ g,
                                float* __restrict__ hn, int total) {
  int i = blockIdx.x * blockDim.x + threadIdx.x;
  if (i >= total) return;
  int n = i / D_HID, d = i % D_HID, hd = d >> 4;
  float gv[S_WALK], m = -1e30f;
#pragma unroll
  for (int s = 0; s < S_WALK; ++s) {
    gv[s] = g[(size_t)s * N_NODES * HEADS + (size_t)n * HEADS + hd];
    m = fmaxf(m, gv[s]);
  }
  float z = 0.f;
#pragma unroll
  for (int s = 0; s < S_WALK; ++s) { gv[s] = __expf(gv[s] - m); z += gv[s]; }
  float acc = 0.f;
#pragma unroll
  for (int s = 0; s < S_WALK; ++s) acc += gv[s] * f[(size_t)s * total + i];
  hn[i] = acc / z;
}

// ---------------- segment softmax pooling ----------------
__global__ void rum_seg_max(const float* __restrict__ gate,
                            const int* __restrict__ seg, unsigned* __restrict__ m,
                            int n) {
  int i = blockIdx.x * blockDim.x + threadIdx.x;
  if (i >= n) return;
  atomicMax(&m[seg[i]], f2s(gate[i]));
}

__global__ void rum_seg_expsum(const float* __restrict__ gate,
                               const int* __restrict__ seg,
                               const unsigned* __restrict__ m,
                               float* __restrict__ e, float* __restrict__ z,
                               int n) {
  int i = blockIdx.x * blockDim.x + threadIdx.x;
  if (i >= n) return;
  float ev = __expf(gate[i] - s2f(m[seg[i]]));
  e[i] = ev;
  atomicAdd(&z[seg[i]], ev);
}

__global__ void rum_seg_wsum(const float* __restrict__ e, const float* __restrict__ z,
                             const int* __restrict__ seg,
                             const float* __restrict__ feat,
                             float* __restrict__ outp, int total) {
  int i = blockIdx.x * blockDim.x + threadIdx.x;
  if (i >= total) return;
  int n = i / D_HID, d = i % D_HID;
  int sg = seg[n];
  float w = e[n] / z[sg];
  atomicAdd(&outp[(size_t)sg * D_HID + d], w * feat[i]);
}

__global__ __launch_bounds__(128)
void rum_fc_out(const float* __restrict__ outp, const float* __restrict__ w1,
                const float* __restrict__ b1, const float* __restrict__ w2,
                const float* __restrict__ b2, const float* __restrict__ loss,
                float* __restrict__ out) {
  __shared__ float row[D_HID];
  __shared__ float red[D_HID];
  int b = blockIdx.x, t = threadIdx.x;
  float x = outp[(size_t)b * D_HID + t];
  row[t] = x > 0.f ? x : __expf(x) - 1.f;          // ELU
  __syncthreads();
  float acc = b1[t];
  for (int k = 0; k < D_HID; ++k) acc += row[k] * w1[(size_t)k * D_HID + t];
  acc = acc > 0.f ? acc : __expf(acc) - 1.f;       // ELU
  red[t] = acc * w2[t];                            // fo_w2 is [D,1]
  __syncthreads();
  for (int s = 64; s; s >>= 1) {
    if (t < s) red[t] += red[t + s];
    __syncthreads();
  }
  if (t == 0) out[b] = red[0] + b2[0];
  if (b == 0 && t == 0) out[B_GRAPH] = *loss;
}

// ---------------- host launch ----------------
extern "C" void kernel_launch(void* const* d_in, const int* in_sizes, int n_in,
                              void* d_out, int out_size, void* d_ws, size_t ws_size,
                              hipStream_t stream) {
  const float* h        = (const float*)d_in[0];
  const int*   walks    = (const int*)d_in[1];
  const int*   n2g      = (const int*)d_in[2];
  const float* fc_in_w  = (const float*)d_in[3];
  const float* fc_in_b  = (const float*)d_in[4];
  const float* gru_wih  = (const float*)d_in[5];
  const float* gru_whh  = (const float*)d_in[6];
  const float* gru_bih  = (const float*)d_in[7];
  const float* gru_bhh  = (const float*)d_in[8];
  const float* dec_w    = (const float*)d_in[9];
  const float* dec_b    = (const float*)d_in[10];
  const float* ln_g     = (const float*)d_in[11];
  const float* ln_b     = (const float*)d_in[12];
  const float* wg_w1    = (const float*)d_in[13];
  const float* wg_b1    = (const float*)d_in[14];
  const float* wg_w2    = (const float*)d_in[15];
  const float* wg_b2    = (const float*)d_in[16];
  const float* wf_w1    = (const float*)d_in[17];
  const float* wf_b1    = (const float*)d_in[18];
  const float* wf_w2    = (const float*)d_in[19];
  const float* wf_b2    = (const float*)d_in[20];
  const float* ag_w1    = (const float*)d_in[21];
  const float* ag_b1    = (const float*)d_in[22];
  const float* ag_w2    = (const float*)d_in[23];
  const float* ag_b2    = (const float*)d_in[24];
  const float* af_w1    = (const float*)d_in[25];
  const float* af_b1    = (const float*)d_in[26];
  const float* af_w2    = (const float*)d_in[27];
  const float* af_b2    = (const float*)d_in[28];
  const float* fo_w1    = (const float*)d_in[29];
  const float* fo_b1    = (const float*)d_in[30];
  const float* fo_w2    = (const float*)d_in[31];
  const float* fo_b2    = (const float*)d_in[32];
  float* out = (float*)d_out;

  // workspace allocator
  char* ws = (char*)d_ws;
  size_t off = 0;
  auto alloc = [&](size_t bytes) -> char* {
    char* p = ws + off;
    off = (off + bytes + 255) & ~(size_t)255;
    return p;
  };
  _Float16* p_fcin = (_Float16*)alloc(16 * 1024);         // K=64  N=128 -> 16 tiles
  _Float16* p_wih0 = (_Float16*)alloc(96 * 1024);         // K=128 N=384 -> 96 tiles
  _Float16* p_wih1 = (_Float16*)alloc(96 * 1024);
  _Float16* p_whh0 = (_Float16*)alloc(96 * 1024);
  _Float16* p_whh1 = (_Float16*)alloc(96 * 1024);
  _Float16* p_dec0 = (_Float16*)alloc(16 * 1024);         // K=128 N=64  -> 16 tiles
  _Float16* p_dec1 = (_Float16*)alloc(16 * 1024);
  _Float16* p_wg1  = (_Float16*)alloc(32 * 1024);         // 128x128 -> 32 tiles
  _Float16* p_wg2  = (_Float16*)alloc(4 * 1024);          // 128x8   -> 4 tiles
  _Float16* p_wf1  = (_Float16*)alloc(32 * 1024);
  _Float16* p_wf2  = (_Float16*)alloc(32 * 1024);
  _Float16* p_ag1  = (_Float16*)alloc(32 * 1024);
  _Float16* p_ag2  = (_Float16*)alloc(4 * 1024);          // 128x1 -> 4 tiles
  _Float16* p_af1  = (_Float16*)alloc(32 * 1024);
  _Float16* p_af2  = (_Float16*)alloc(32 * 1024);
  float* h_cur  = (float*)alloc((size_t)N_NODES * D_HID * 4);
  float* state  = (float*)alloc((size_t)SN * D_HID * 4);
  float* fbuf   = (float*)alloc((size_t)SN * D_HID * 4);
  float* fbuf2  = (float*)alloc((size_t)SN * D_HID * 4);
  float* gbuf   = (float*)alloc((size_t)SN * HEADS * 4);
  float* hn     = (float*)alloc((size_t)N_NODES * D_HID * 4);
  float* gatev  = (float*)alloc((size_t)N_NODES * 4);
  float* ebuf   = (float*)alloc((size_t)N_NODES * 4);
  unsigned* mbuf = (unsigned*)alloc(B_GRAPH * 4);
  float* zbuf   = (float*)alloc(B_GRAPH * 4);
  float* outpool = (float*)alloc((size_t)B_GRAPH * D_HID * 4);
  float* lossb  = (float*)alloc(256);

  // 0) init accumulators
  rum_init<<<(B_GRAPH * D_HID + 255) / 256, 256, 0, stream>>>(lossb, mbuf, zbuf,
                                                              outpool);
  // 1) pack weights to f16 WMMA-B layout
  rum_pack_b<<<64, 256, 0, stream>>>(fc_in_w, p_fcin, 64, 128);
  rum_pack_b<<<64, 256, 0, stream>>>(gru_wih, p_wih0, 128, 384);
  rum_pack_b<<<64, 256, 0, stream>>>(gru_wih + 128 * 384, p_wih1, 128, 384);
  rum_pack_b<<<64, 256, 0, stream>>>(gru_whh, p_whh0, 128, 384);
  rum_pack_b<<<64, 256, 0, stream>>>(gru_whh + 128 * 384, p_whh1, 128, 384);
  rum_pack_b<<<64, 256, 0, stream>>>(dec_w, p_dec0, 128, 64);
  rum_pack_b<<<64, 256, 0, stream>>>(dec_w + 128 * 64, p_dec1, 128, 64);
  rum_pack_b<<<64, 256, 0, stream>>>(wg_w1, p_wg1, 128, 128);
  rum_pack_b<<<64, 256, 0, stream>>>(wg_w2, p_wg2, 128, 8);
  rum_pack_b<<<64, 256, 0, stream>>>(wf_w1, p_wf1, 128, 128);
  rum_pack_b<<<64, 256, 0, stream>>>(wf_w2, p_wf2, 128, 128);
  rum_pack_b<<<64, 256, 0, stream>>>(ag_w1, p_ag1, 128, 128);
  rum_pack_b<<<64, 256, 0, stream>>>(ag_w2, p_ag2, 128, 1);
  rum_pack_b<<<64, 256, 0, stream>>>(af_w1, p_af1, 128, 128);
  rum_pack_b<<<64, 256, 0, stream>>>(af_w2, p_af2, 128, 128);

  // 2) fc_in: h_cur = h @ fc_in_w + b   (K=64)
  rum_gemm<2, 128, 0><<<(N_NODES / 16 + 3) / 4, 128, 16 * 1024, stream>>>(
      h, p_fcin, fc_in_b, h_cur, N_NODES);

  const float lscale = SS_W / (float)((size_t)SN * F_IN);
  // 192 KB weights + 8 waves * 8 KB transpose scratch = 256 KB (<= 320 KB/WGP)
  const size_t gru_smem = (size_t)2 * 96 * 1024 + 8 * 16 * D_HID * 4;

  // 3) layer 0 GRU scan (fused, WMMA); 8 waves/block -> 2 waves/SIMD
  rum_gru_layer<<<SN / 128, 256, gru_smem, stream>>>(
      h_cur, walks, p_wih0, p_whh0, gru_bih, gru_bhh, p_dec0, dec_b, h, state,
      lossb, lscale);
  // 4) hh = silu(hT).mean(0)
  rum_silu_mean<<<(N_NODES * D_HID + 255) / 256, 256, 0, stream>>>(
      state, h_cur, N_NODES * D_HID);
  // 5) layer 1 GRU scan
  rum_gru_layer<<<SN / 128, 256, gru_smem, stream>>>(
      h_cur, walks + (size_t)SN * L_WALK, p_wih1, p_whh1, gru_bih + 384,
      gru_bhh + 384, p_dec1, dec_b + 64, h, state, lossb, lscale);

  // 6) LayerNorm in-place on state [S*N, D]
  rum_layernorm<<<SN / 8, 256, 0, stream>>>(state, ln_g, ln_b, SN);

  // 7) MHA: f = mlp2(hh; wf), g = mlp2(hh; wg), softmax over S, combine
  rum_gemm<4, 128, 1><<<SN / 64, 128, 32 * 1024, stream>>>(state, p_wf1, wf_b1,
                                                           fbuf, SN);
  rum_gemm<4, 128, 0><<<SN / 64, 128, 32 * 1024, stream>>>(fbuf, p_wf2, wf_b2,
                                                           fbuf2, SN);
  rum_gemm<4, 128, 1><<<SN / 64, 128, 32 * 1024, stream>>>(state, p_wg1, wg_b1,
                                                           fbuf, SN);
  rum_gemm<4, 8, 0><<<SN / 64, 128, 4 * 1024, stream>>>(fbuf, p_wg2, wg_b2, gbuf,
                                                        SN);
  rum_mha_combine<<<(N_NODES * D_HID + 255) / 256, 256, 0, stream>>>(
      fbuf2, gbuf, hn, N_NODES * D_HID);

  // 8) GlobalAttentionPooling
  rum_gemm<4, 128, 1><<<(N_NODES / 16 + 3) / 4, 128, 32 * 1024, stream>>>(
      hn, p_ag1, ag_b1, fbuf, N_NODES);
  rum_gemm<4, 1, 0><<<(N_NODES / 16 + 3) / 4, 128, 4 * 1024, stream>>>(
      fbuf, p_ag2, ag_b2, gatev, N_NODES);
  rum_gemm<4, 128, 1><<<(N_NODES / 16 + 3) / 4, 128, 32 * 1024, stream>>>(
      hn, p_af1, af_b1, fbuf, N_NODES);
  rum_gemm<4, 128, 0><<<(N_NODES / 16 + 3) / 4, 128, 32 * 1024, stream>>>(
      fbuf, p_af2, af_b2, fbuf2, N_NODES);
  rum_seg_max<<<(N_NODES + 255) / 256, 256, 0, stream>>>(gatev, n2g, mbuf, N_NODES);
  rum_seg_expsum<<<(N_NODES + 255) / 256, 256, 0, stream>>>(gatev, n2g, mbuf, ebuf,
                                                            zbuf, N_NODES);
  rum_seg_wsum<<<(N_NODES * D_HID + 255) / 256, 256, 0, stream>>>(
      ebuf, zbuf, n2g, fbuf2, outpool, N_NODES * D_HID);

  // 9) fc_out + final loss write
  rum_fc_out<<<B_GRAPH, 128, 0, stream>>>(outpool, fo_w1, fo_b1, fo_w2, fo_b2,
                                          lossb, out);
  (void)in_sizes; (void)n_in; (void)out_size; (void)ws_size;
}
